// S4_Layer_41721312313628
// MI455X (gfx1250) — compile-verified, matrix operand order checked by reading, not run on hardware
//
#include <hip/hip_runtime.h>
#include <math.h>

#define Bn 8
#define Ln 2048
#define Hn 512
#define Nst 64
#define MR (Bn * Ln)        /* 16384 rows */
#define HN (Hn * Nst)       /* 32768 */

typedef __attribute__((ext_vector_type(16))) __bf16 v16bf;
typedef __attribute__((ext_vector_type(8)))  float  v8f;

union BF16Frag { unsigned u[8]; v16bf v; };

static __device__ __forceinline__ __bf16 f2bf(float f) { return (__bf16)f; }

// CDNA5 async global->LDS copy (16B per lane), tracked with ASYNCcnt.
// VDST = LDS byte address (low 32 bits of the generic shared pointer),
// VADDR = 64-bit global address, GV mode (saddr = off).
static __device__ __forceinline__ void async_copy16(const __bf16* g, __bf16* l) {
  unsigned lds = (unsigned)(uintptr_t)l;
  asm volatile("global_load_async_to_lds_b128 %0, %1, off"
               :: "v"(lds), "v"(g) : "memory");
}
static __device__ __forceinline__ void wait_async0() {
  asm volatile("s_wait_asynccnt 0x0" ::: "memory");
}

// ---------------------------------------------------------------------------
// 1) Convert weights to bf16 (W_out 1024x512, W1 512x512, W2 512x512),
//    all already stored [n][k] row-major as needed by the GEMM B operand.
// ---------------------------------------------------------------------------
#define WOUT_E (1024 * 512)
#define W1_E   (512 * 512)
#define W2_E   (512 * 512)

__global__ void cvt_weights_kernel(const float* __restrict__ Wout,
                                   const float* __restrict__ W1,
                                   const float* __restrict__ W2,
                                   __bf16* __restrict__ woutb,
                                   __bf16* __restrict__ w1b,
                                   __bf16* __restrict__ w2b) {
  int i = blockIdx.x * blockDim.x + threadIdx.x;
  if (i < WOUT_E) {
    woutb[i] = f2bf(Wout[i]);
  } else if (i < WOUT_E + W1_E) {
    int j = i - WOUT_E;
    w1b[j] = f2bf(W1[j]);
  } else if (i < WOUT_E + W1_E + W2_E) {
    int j = i - WOUT_E - W1_E;
    w2b[j] = f2bf(W2[j]);
  }
}

// ---------------------------------------------------------------------------
// 2) ZOH discretization: w = exp(dt*A), Ct = C * (w - 1)/A   (complex, per h,n)
//    par layout: [0]=wr [1]=wi [2]=ctr [3]=cti, each HN floats.
// ---------------------------------------------------------------------------
__global__ void precompute_kernel(const float* __restrict__ log_dt,
                                  const float* __restrict__ A_re,
                                  const float* __restrict__ A_im,
                                  const float* __restrict__ C_re,
                                  const float* __restrict__ C_im,
                                  float* __restrict__ par) {
  int i = blockIdx.x * blockDim.x + threadIdx.x;
  if (i >= HN) return;
  int h = i / Nst;
  float dt = expf(log_dt[h]);
  float ar = A_re[i], ai = A_im[i];
  float er = expf(dt * ar);
  float wr = er * cosf(dt * ai);
  float wi = er * sinf(dt * ai);
  float nr = wr - 1.0f, ni = wi;
  float den = ar * ar + ai * ai;
  float rden = 1.0f / den;
  float qr = (nr * ar + ni * ai) * rden;
  float qi = (ni * ar - nr * ai) * rden;
  float cr = C_re[i], ci = C_im[i];
  par[0 * HN + i] = wr;
  par[1 * HN + i] = wi;
  par[2 * HN + i] = cr * qr - ci * qi;
  par[3 * HN + i] = cr * qi + ci * qr;
}

// ---------------------------------------------------------------------------
// 3) SSM recurrence scan + D-skip + GELU, output bf16 rows (b*L+l, h).
//    One block per (b,h); 64 threads, one diagonal state per thread.
//    Chunked over 16 timesteps; LDS tree reduction over the 64 states.
// ---------------------------------------------------------------------------
__global__ __launch_bounds__(64) void ssm_scan_kernel(
    const float* __restrict__ x, const float* __restrict__ par,
    const float* __restrict__ Dskip, __bf16* __restrict__ y1) {
  int bh = blockIdx.x;
  int b = bh / Hn, h = bh % Hn;
  int n = threadIdx.x;

  float wr  = par[0 * HN + h * Nst + n];
  float wi  = par[1 * HN + h * Nst + n];
  float ctr = par[2 * HN + h * Nst + n];
  float cti = par[3 * HN + h * Nst + n];
  float D = Dskip[h];

  __shared__ float ubuf[16];
  __shared__ float cbuf[64 * 17];   // [state][pos(0..15 per chunk, padded)]
  __shared__ float part[16 * 4];

  float sr = 0.0f, si = 0.0f;
  const float* xb = x + (size_t)b * Ln * Hn + h;

  for (int l0 = 0; l0 < Ln; l0 += 16) {
    if (n < 16) ubuf[n] = xb[(size_t)(l0 + n) * Hn];
    __syncthreads();
#pragma unroll
    for (int j = 0; j < 16; ++j) {
      float u = ubuf[j];
      float nsr = fmaf(wr, sr, fmaf(-wi, si, u));
      float nsi = fmaf(wr, si, wi * sr);
      sr = nsr; si = nsi;
      cbuf[n * 17 + j] = 2.0f * (ctr * sr - cti * si);
    }
    __syncthreads();
    {
      int j = n & 15, q = n >> 4;
      float s = 0.0f;
#pragma unroll
      for (int t = 0; t < 16; ++t) s += cbuf[(q * 16 + t) * 17 + j];
      part[j * 4 + q] = s;
    }
    __syncthreads();
    if (n < 16) {
      float y = part[n * 4 + 0] + part[n * 4 + 1] + part[n * 4 + 2] + part[n * 4 + 3];
      y += D * ubuf[n];
      y = 0.5f * y * (1.0f + erff(y * 0.70710678f));   // exact GELU
      y1[((size_t)b * Ln + l0 + n) * Hn + h] = f2bf(y);
    }
    __syncthreads();
  }
}

// ---------------------------------------------------------------------------
// 4) bf16 WMMA GEMM: C[M,N] (f32) = A[M,K] (bf16, row-major)
//                                 x Bw[N,K]^T (bf16, row-major = weight [n][k])
//    Block tile 128x128, K-step 32, 8 waves (4x2), wave tile 32x64,
//    8x v_wmma_f32_16x16x32_bf16 per wave per K-step.
//    Double-buffered LDS fed by global_load_async_to_lds_b128 (ASYNCcnt);
//    K-chunk i+1 copies overlap WMMA on chunk i; one barrier per K-step.
// ---------------------------------------------------------------------------
__global__ __launch_bounds__(256) void gemm_bf16_wmma_kernel(
    const __bf16* __restrict__ A, const __bf16* __restrict__ Bw,
    float* __restrict__ C, int Mdim, int Ndim, int Kdim) {
  __shared__ __attribute__((aligned(16))) __bf16 sA[2][128 * 32];
  __shared__ __attribute__((aligned(16))) __bf16 sB[2][128 * 32];

  int tid = threadIdx.x;
  int lane = tid & 31, w = tid >> 5;
  int wm = w >> 1, wn = w & 1;
  int tilesN = Ndim >> 7;
  int bm = blockIdx.x / tilesN, bn = blockIdx.x % tilesN;
  int m0 = bm * 128, n0 = bn * 128;

  // staging slots for this thread: 2 x 16B per operand per K-step
  const int r0 = tid >> 2,          c0 = (tid & 3) * 8;
  const int r1 = (tid + 256) >> 2,  c1 = (tid & 3) * 8;   // (tid+256)&3 == tid&3

#define STAGE(buf, k0)                                                         \
  do {                                                                         \
    async_copy16(&A[(size_t)(m0 + r0) * Kdim + (k0) + c0], &sA[buf][r0 * 32 + c0]); \
    async_copy16(&A[(size_t)(m0 + r1) * Kdim + (k0) + c1], &sA[buf][r1 * 32 + c1]); \
    async_copy16(&Bw[(size_t)(n0 + r0) * Kdim + (k0) + c0], &sB[buf][r0 * 32 + c0]); \
    async_copy16(&Bw[(size_t)(n0 + r1) * Kdim + (k0) + c1], &sB[buf][r1 * 32 + c1]); \
  } while (0)

  v8f acc[2][4];
#pragma unroll
  for (int i = 0; i < 2; ++i)
#pragma unroll
    for (int j = 0; j < 4; ++j) acc[i][j] = (v8f){0,0,0,0,0,0,0,0};

  const int mr = lane & 15;
  const int kb = (lane >> 4) * 8;
  int koff[8];
#pragma unroll
  for (int j = 0; j < 8; ++j) koff[j] = kb + ((j < 4) ? (2 * j) : (16 + 2 * (j - 4)));

  const int iters = Kdim >> 5;
  STAGE(0, 0);                               // prologue: fill buffer 0

  for (int i = 0; i < iters; ++i) {
    wait_async0();                           // buf (i&1) resident in LDS
    __syncthreads();                         // all waves done reading buf (i+1)&1
    if (i + 1 < iters) STAGE((i + 1) & 1, (i + 1) * 32);   // overlap with WMMA

    const __bf16* la = sA[i & 1];
    const __bf16* lb = sB[i & 1];

    BF16Frag af[2], bf[4];
#pragma unroll
    for (int mt = 0; mt < 2; ++mt) {
      int m = wm * 32 + mt * 16 + mr;
#pragma unroll
      for (int j = 0; j < 8; ++j)
        af[mt].u[j] = *(const unsigned*)&la[m * 32 + koff[j]];
    }
#pragma unroll
    for (int nt = 0; nt < 4; ++nt) {
      int nn = wn * 64 + nt * 16 + mr;
#pragma unroll
      for (int j = 0; j < 8; ++j)
        bf[nt].u[j] = *(const unsigned*)&lb[nn * 32 + koff[j]];
    }
#pragma unroll
    for (int mt = 0; mt < 2; ++mt)
#pragma unroll
      for (int nt = 0; nt < 4; ++nt)
        acc[mt][nt] = __builtin_amdgcn_wmma_f32_16x16x32_bf16(
            false, af[mt].v, false, bf[nt].v, (short)0, acc[mt][nt],
            false, false);
  }
#undef STAGE

  // D layout: VGPR r holds row r + 8*(lane>=16), col = lane&15
#pragma unroll
  for (int mt = 0; mt < 2; ++mt)
#pragma unroll
    for (int nt = 0; nt < 4; ++nt)
#pragma unroll
      for (int r = 0; r < 8; ++r) {
        int m = m0 + wm * 32 + mt * 16 + r + 8 * (lane >> 4);
        int nn = n0 + wn * 64 + nt * 16 + (lane & 15);
        C[(size_t)m * Ndim + nn] = acc[mt][nt][r];
      }
}

// ---------------------------------------------------------------------------
// 5) Fused bias + GLU + residual(x) + LayerNorm(g1,b1); emit f32 + bf16.
//    One block per row (256 thr, 2 cols/thread over H=512).
// ---------------------------------------------------------------------------
__global__ __launch_bounds__(256) void glu_res_ln_kernel(
    const float* __restrict__ Z, const float* __restrict__ b_out,
    const float* __restrict__ x, const float* __restrict__ g1,
    const float* __restrict__ be1, float* __restrict__ x1f,
    __bf16* __restrict__ x1b) {
  int r = blockIdx.x, t = threadIdx.x;
  __shared__ float red[256];
  float v[2];
  float lsum = 0.0f;
#pragma unroll
  for (int i = 0; i < 2; ++i) {
    int c = t + i * 256;
    float a = Z[(size_t)r * 1024 + c] + b_out[c];
    float g = Z[(size_t)r * 1024 + 512 + c] + b_out[512 + c];
    float y2 = a * (1.0f / (1.0f + expf(-g)));
    v[i] = x[(size_t)r * Hn + c] + y2;
    lsum += v[i];
  }
  red[t] = lsum; __syncthreads();
  for (int s = 128; s >= 1; s >>= 1) { if (t < s) red[t] += red[t + s]; __syncthreads(); }
  float mean = red[0] * (1.0f / Hn);
  __syncthreads();
  float lv = 0.0f;
#pragma unroll
  for (int i = 0; i < 2; ++i) { float d = v[i] - mean; lv += d * d; }
  red[t] = lv; __syncthreads();
  for (int s = 128; s >= 1; s >>= 1) { if (t < s) red[t] += red[t + s]; __syncthreads(); }
  float inv = rsqrtf(red[0] * (1.0f / Hn) + 1e-5f);
#pragma unroll
  for (int i = 0; i < 2; ++i) {
    int c = t + i * 256;
    float o = (v[i] - mean) * inv * g1[c] + be1[c];
    x1f[(size_t)r * Hn + c] = o;
    x1b[(size_t)r * Hn + c] = f2bf(o);
  }
}

// ---------------------------------------------------------------------------
// 6) bias + ReLU -> bf16 (FFN hidden)
// ---------------------------------------------------------------------------
__global__ void bias_relu_kernel(const float* __restrict__ T1,
                                 const float* __restrict__ bc1,
                                 __bf16* __restrict__ hid) {
  size_t idx = (size_t)blockIdx.x * blockDim.x + threadIdx.x;
  if (idx >= (size_t)MR * Hn) return;
  int c = (int)(idx & (Hn - 1));
  float v = T1[idx] + bc1[c];
  hid[idx] = f2bf(fmaxf(v, 0.0f));
}

// ---------------------------------------------------------------------------
// 7) Fused residual(x1) + bias + LayerNorm(g2,b2) -> d_out (f32)
// ---------------------------------------------------------------------------
__global__ __launch_bounds__(256) void final_res_ln_kernel(
    const float* __restrict__ x1f, const float* __restrict__ T2,
    const float* __restrict__ bc2, const float* __restrict__ g2,
    const float* __restrict__ be2, float* __restrict__ out) {
  int r = blockIdx.x, t = threadIdx.x;
  __shared__ float red[256];
  float v[2];
  float lsum = 0.0f;
#pragma unroll
  for (int i = 0; i < 2; ++i) {
    int c = t + i * 256;
    v[i] = x1f[(size_t)r * Hn + c] + T2[(size_t)r * Hn + c] + bc2[c];
    lsum += v[i];
  }
  red[t] = lsum; __syncthreads();
  for (int s = 128; s >= 1; s >>= 1) { if (t < s) red[t] += red[t + s]; __syncthreads(); }
  float mean = red[0] * (1.0f / Hn);
  __syncthreads();
  float lv = 0.0f;
#pragma unroll
  for (int i = 0; i < 2; ++i) { float d = v[i] - mean; lv += d * d; }
  red[t] = lv; __syncthreads();
  for (int s = 128; s >= 1; s >>= 1) { if (t < s) red[t] += red[t + s]; __syncthreads(); }
  float inv = rsqrtf(red[0] * (1.0f / Hn) + 1e-5f);
#pragma unroll
  for (int i = 0; i < 2; ++i) {
    int c = t + i * 256;
    out[(size_t)r * Hn + c] = (v[i] - mean) * inv * g2[c] + be2[c];
  }
}

// ---------------------------------------------------------------------------
// Host launcher
// ---------------------------------------------------------------------------
extern "C" void kernel_launch(void* const* d_in, const int* in_sizes, int n_in,
                              void* d_out, int out_size, void* d_ws, size_t ws_size,
                              hipStream_t stream) {
  const float* x      = (const float*)d_in[0];
  const float* log_dt = (const float*)d_in[1];
  const float* A_re   = (const float*)d_in[2];
  const float* A_im   = (const float*)d_in[3];
  const float* C_re   = (const float*)d_in[4];
  const float* C_im   = (const float*)d_in[5];
  const float* D_skip = (const float*)d_in[6];
  const float* W_out  = (const float*)d_in[7];
  const float* b_out  = (const float*)d_in[8];
  const float* g1     = (const float*)d_in[9];
  const float* be1    = (const float*)d_in[10];
  const float* g2     = (const float*)d_in[11];
  const float* be2    = (const float*)d_in[12];
  const float* W1     = (const float*)d_in[13];
  const float* bc1    = (const float*)d_in[14];
  const float* W2     = (const float*)d_in[15];
  const float* bc2    = (const float*)d_in[16];
  float* out = (float*)d_out;

  unsigned char* ws = (unsigned char*)d_ws;
  // workspace layout (256B-aligned offsets)
  float*  par   = (float*)(ws + 0);                       // 4*HN f32       = 512 KB
  __bf16* woutb = (__bf16*)(ws + 524288);                 // 1024x512 bf16  = 1 MB
  __bf16* w1b   = (__bf16*)(ws + 1572864);                // 512x512 bf16
  __bf16* w2b   = (__bf16*)(ws + 2097152);                // 512x512 bf16
  __bf16* y1b   = (__bf16*)(ws + 2621440);                // 16384x512 bf16 = 16 MB
  float*  Z     = (float*)(ws + 19398656);                // 16384x1024 f32 = 64 MB
  float*  T1    = Z;                                      // reuse (32 MB)
  float*  T2    = (float*)(ws + 19398656 + 33554432);     // reuse upper half
  float*  x1f   = (float*)(ws + 86507520);                // 16384x512 f32  = 32 MB
  __bf16* x1b   = (__bf16*)(ws + 120061952);              // 16384x512 bf16 = 16 MB
  __bf16* hidb  = (__bf16*)(ws + 136839168);              // 16384x512 bf16 = 16 MB

  // 1) weights -> bf16
  cvt_weights_kernel<<<(WOUT_E + W1_E + W2_E + 255) / 256, 256, 0, stream>>>(
      W_out, W1, W2, woutb, w1b, w2b);
  // 2) discretization
  precompute_kernel<<<(HN + 255) / 256, 256, 0, stream>>>(log_dt, A_re, A_im,
                                                          C_re, C_im, par);
  // 3) SSM scan + GELU -> y1 (bf16)
  ssm_scan_kernel<<<Bn * Hn, 64, 0, stream>>>(x, par, D_skip, y1b);
  // 4) Z = y1 @ W_out^T   (16384x512 x 512x1024)
  gemm_bf16_wmma_kernel<<<(MR / 128) * (1024 / 128), 256, 0, stream>>>(
      y1b, woutb, Z, MR, 1024, Hn);
  // 5) GLU + residual + LN1 -> x1 (f32 + bf16)
  glu_res_ln_kernel<<<MR, 256, 0, stream>>>(Z, b_out, x, g1, be1, x1f, x1b);
  // 6) T1 = x1 @ W1^T
  gemm_bf16_wmma_kernel<<<(MR / 128) * (Hn / 128), 256, 0, stream>>>(
      x1b, w1b, T1, MR, Hn, Hn);
  // 7) hidden = relu(T1 + bc1) -> bf16
  bias_relu_kernel<<<(int)(((size_t)MR * Hn + 255) / 256), 256, 0, stream>>>(
      T1, bc1, hidb);
  // 8) T2 = hidden @ W2^T
  gemm_bf16_wmma_kernel<<<(MR / 128) * (Hn / 128), 256, 0, stream>>>(
      hidb, w2b, T2, MR, Hn, Hn);
  // 9) out = LN2(x1 + T2 + bc2)
  final_res_ln_kernel<<<MR, 256, 0, stream>>>(x1f, T2, bc2, g2, be2, out);
}
